// CrossAttentionLayer_13700945674702
// MI455X (gfx1250) — matrix operand order repaired and checked by reading, not compiled, not run on hardware
//
#include <hip/hip_runtime.h>
#include <hip/hip_bf16.h>
#include <math.h>

// ---------------------------------------------------------------------------
// CrossAttentionLayer on MI455X (gfx1250, wave32, WMMA).
// bf16 storage + v_wmma_f32_16x16x32_bf16 (f32 accumulate) for all GEMMs;
// f32 softmax / LayerNorm. Most global tile/fragment loads are 128-bit vectorized;
// the GEMM A-tile is staged via the Tensor Data Mover (tensor_load_to_lds,
// s_wait_tensorcnt) when the builtin is available.
// ---------------------------------------------------------------------------

typedef __attribute__((ext_vector_type(16))) __bf16 v16bf;
typedef __attribute__((ext_vector_type(8)))  float  v8f;
typedef __attribute__((ext_vector_type(4)))  unsigned int v4u;
typedef __attribute__((ext_vector_type(8)))  int v8i;
typedef __attribute__((ext_vector_type(4)))  int v4i;

#define D_MODEL 768
#define D_FF    3072
#define N_HEADS 12
#define DK      64
#define BB      8
#define LM      512
#define LP      1024

#define HAVE_TDM (__has_builtin(__builtin_amdgcn_tensor_load_to_lds) && \
                  __has_builtin(__builtin_amdgcn_s_wait_tensorcnt))

// ---- WMMA fragment loaders -------------------------------------------------
// A-matrix 16x32 bf16 layout: lanes 0-15 m=lane, k-pairs {0..7,16..23};
// lanes 16-31 k-pairs {8..15,24..31}. As u32 words: lanes<16 -> words 0-3,8-11;
// lanes>=16 -> words 4-7,12-15. Row pointer must be 16B aligned.
__device__ inline v16bf load_frag_a(const __bf16* row, int lane) {
  union { v16bf v; uint4 x[2]; } f;
  const uint4* p = (const uint4*)row;
  int i0 = (lane < 16) ? 0 : 1;
  f.x[0] = p[i0];
  f.x[1] = p[i0 + 2];
  return f.v;
}
// B-matrix 32x16 bf16 layout (row = the N-index line of 32 k-values):
// lanes 0-15 K={0..7,16..23}, lanes 16-31 K={8..15,24..31} (same as A).
__device__ inline v16bf load_frag_b(const __bf16* row, int lane) {
  union { v16bf v; uint4 x[2]; } f;
  const uint4* p = (const uint4*)row;
  int i0 = (lane < 16) ? 0 : 1;
  f.x[0] = p[i0];
  f.x[1] = p[i0 + 2];
  return f.v;
}

// ---------------------------------------------------------------- f32 -> bf16
__global__ __launch_bounds__(256) void cvt_bf16_kernel(const float* __restrict__ s,
                                                       __bf16* __restrict__ d, int n4) {
  int i = blockIdx.x * 256 + threadIdx.x;
  if (i < n4) {
    float4 f = ((const float4*)s)[i];
    union { __bf16 h[4]; uint2 u; } o;
    o.h[0] = (__bf16)f.x; o.h[1] = (__bf16)f.y;
    o.h[2] = (__bf16)f.z; o.h[3] = (__bf16)f.w;
    ((uint2*)d)[i] = o.u;
  }
}

// ------------------------------------------------------- generic bf16 WMMA GEMM
// C[M,N] = A[M,K] * B[K,N] (+bias) (gelu?) ; A,B bf16 row-major.
// Tile 64x64x32; 256 threads = 8 waves; wave w: m-sub (w&3), n-half (w>>2).
__global__ __launch_bounds__(256) void gemm_bf16_kernel(
    const __bf16* __restrict__ A, const __bf16* __restrict__ B,
    int M, int N, int K,
    const float* __restrict__ bias, int act,
    float* __restrict__ outF, __bf16* __restrict__ outB) {
  __shared__ __bf16 sA[64 * 32];   // [m][k]
  __shared__ __bf16 sBt[64 * 32];  // [n][k]

  const int tid  = threadIdx.x;
  const int lane = tid & 31;
  const int wave = tid >> 5;
  const int msub = wave & 3;
  const int nhalf = wave >> 2;
  const int tile_m = blockIdx.y * 64;
  const int tile_n = blockIdx.x * 64;

  v8f acc0 = {0.f,0.f,0.f,0.f,0.f,0.f,0.f,0.f};
  v8f acc1 = {0.f,0.f,0.f,0.f,0.f,0.f,0.f,0.f};

  for (int k0 = 0; k0 < K; k0 += 32) {
    __syncthreads();
#if HAVE_TDM
    // Tensor Data Mover: DMA the 64x32 bf16 A tile (row stride K) into sA.
    if (wave == 0) {
      unsigned lds_off = (unsigned)(unsigned long long)(void*)sA;
      unsigned long long ga =
          (unsigned long long)(const void*)(A + (size_t)tile_m * K + k0);
      v4u g0;
      g0[0] = 1u;                                   // count=1 user descriptor
      g0[1] = lds_off;                              // lds_addr
      g0[2] = (unsigned)ga;                         // global_addr[31:0]
      g0[3] = (unsigned)((ga >> 32) & 0x01FFFFFFu)  // global_addr[56:32]
              | 0x80000000u;                        // type=2 ("image")
      v8i g1;
      g1[0] = (int)(1u << 16);         // data_size=1 (2 bytes), mask=0
      g1[1] = (int)(32u << 16);        // tensor_dim0 = 32
      g1[2] = (int)(64u << 16);        // tensor_dim1 = 64
      g1[3] = (int)(32u << 16);        // tile_dim0 = 32
      g1[4] = 64;                      // tile_dim1 = 64, tile_dim2 = 0
      g1[5] = (int)(unsigned)K;        // tensor_dim0_stride = K (elements)
      g1[6] = 0;
      g1[7] = 0;
      v4i z = {0, 0, 0, 0};
#if __clang_major__ >= 23
      v8i z8 = {0, 0, 0, 0, 0, 0, 0, 0};
      __builtin_amdgcn_tensor_load_to_lds(g0, g1, z, z, z8, 0);
#else
      __builtin_amdgcn_tensor_load_to_lds(g0, g1, z, z, 0);
#endif
    }
#else
    {  // manual vectorized A staging: one b128 per thread
      int r = tid >> 2;
      int c = (tid & 3) * 8;
      *(uint4*)(sA + r * 32 + c) =
          *(const uint4*)(A + (size_t)(tile_m + r) * K + k0 + c);
    }
#endif
    {  // B staged transposed [n][k]: b128 global load, 8 b16 LDS stores
      int kk = tid >> 3;
      int n0 = (tid & 7) * 8;
      union { uint4 u; __bf16 h[8]; } v;
      v.u = *(const uint4*)(B + (size_t)(k0 + kk) * N + tile_n + n0);
#pragma unroll
      for (int j = 0; j < 8; ++j) sBt[(n0 + j) * 32 + kk] = v.h[j];
    }
    if (k0 + 32 < K)  // pull next B slab toward L2 (global_prefetch_b8)
      __builtin_prefetch(B + (size_t)(k0 + 32 + (tid >> 3)) * N + tile_n, 0, 1);
#if HAVE_TDM
    if (wave == 0) __builtin_amdgcn_s_wait_tensorcnt(0);
#endif
    __syncthreads();

    v16bf afrag = load_frag_a(sA + (msub * 16 + (lane & 15)) * 32, lane);
    v16bf b0 = load_frag_b(sBt + (nhalf * 32 + (lane & 15)) * 32, lane);
    v16bf b1 = load_frag_b(sBt + (nhalf * 32 + 16 + (lane & 15)) * 32, lane);
    acc0 = __builtin_amdgcn_wmma_f32_16x16x32_bf16(false, afrag, false, b0,
                                                   (short)0, acc0, false, false);
    acc1 = __builtin_amdgcn_wmma_f32_16x16x32_bf16(false, afrag, false, b1,
                                                   (short)0, acc1, false, false);
  }

  // Epilogue. C layout: VGPR q -> M = q (+8 for lanes 16-31), N = lane&15.
  const int mrow0 = tile_m + msub * 16 + ((lane < 16) ? 0 : 8);
  const int ncol0 = lane & 15;
#pragma unroll
  for (int q = 0; q < 8; ++q) {
    int row = mrow0 + q;
#pragma unroll
    for (int t = 0; t < 2; ++t) {
      int col = tile_n + nhalf * 32 + t * 16 + ncol0;
      float v = (t == 0) ? acc0[q] : acc1[q];
      if (bias) v += bias[col];
      if (act == 1) v = 0.5f * v * (1.0f + erff(v * 0.70710678118f));  // exact gelu
      if (outF) outF[(size_t)row * N + col] = v;
      if (outB) outB[(size_t)row * N + col] = (__bf16)v;
    }
  }
}

// --------------------------------------------------- attention: masked scores
// grid (LM/16, H, B), block 128 (4 waves x 256 keys). Writes scaled+masked
// scores into attnW (the attn_weights region of d_out, softmaxed next).
__global__ __launch_bounds__(128) void attn_scores_kernel(
    const __bf16* __restrict__ Qp, const __bf16* __restrict__ Kp,
    const unsigned char* __restrict__ mask, float* __restrict__ attnW) {
  const int b = blockIdx.z, h = blockIdx.y, q0 = blockIdx.x * 16;
  const int tid = threadIdx.x, lane = tid & 31, wave = tid >> 5;

  __shared__ int allm;
  if (tid == 0) allm = 1;
  __syncthreads();
  const unsigned char* mrow = mask + b * LP;
  {
    int any0 = 0;
    for (int i = tid; i < LP; i += 128) any0 |= (mrow[i] == 0);
    if (any0) atomicAnd(&allm, 0);
  }
  __syncthreads();
  const int am = allm;

  // Q fragments (16 rows x 64 dk = two k-steps) straight from global (b128).
  const __bf16* Qrow = Qp + (size_t)(b * LM + q0 + (lane & 15)) * D_MODEL + h * DK;
  v16bf aq0 = load_frag_a(Qrow, lane);
  v16bf aq1 = load_frag_a(Qrow + 32, lane);

  const __bf16* Kbase = Kp + (size_t)b * LP * D_MODEL + h * DK;
  float* wbase = attnW + ((size_t)(b * N_HEADS + h) * LM + q0) * LP;

  for (int nt = 0; nt < 16; ++nt) {
    int n0  = wave * 256 + nt * 16;
    int key = n0 + (lane & 15);
    // scores B-matrix element (kdim=d, n=key) = K[key][d]: d is contiguous.
    const __bf16* Krow = Kbase + (size_t)key * D_MODEL;
    v16bf bk0 = load_frag_b(Krow, lane);
    v16bf bk1 = load_frag_b(Krow + 32, lane);
    v8f acc = {0.f,0.f,0.f,0.f,0.f,0.f,0.f,0.f};
    acc = __builtin_amdgcn_wmma_f32_16x16x32_bf16(false, aq0, false, bk0,
                                                  (short)0, acc, false, false);
    acc = __builtin_amdgcn_wmma_f32_16x16x32_bf16(false, aq1, false, bk1,
                                                  (short)0, acc, false, false);
    bool msk = (mrow[key] != 0) && !(am && key == 0);  // all-masked guard
#pragma unroll
    for (int q = 0; q < 8; ++q) {
      int m = q + ((lane < 16) ? 0 : 8);
      wbase[(size_t)m * LP + key] = msk ? -3.0e38f : acc[q] * 0.125f;
    }
  }
}

// ------------------------------------------------------ in-place row softmax
__global__ __launch_bounds__(256) void softmax_kernel(float* __restrict__ w) {
  float* p = w + (size_t)blockIdx.x * LP;
  __shared__ float red[256];
  int tid = threadIdx.x;
  float v0 = p[tid], v1 = p[tid + 256], v2 = p[tid + 512], v3 = p[tid + 768];
  red[tid] = fmaxf(fmaxf(v0, v1), fmaxf(v2, v3));
  __syncthreads();
  for (int st = 128; st > 0; st >>= 1) {
    if (tid < st) red[tid] = fmaxf(red[tid], red[tid + st]);
    __syncthreads();
  }
  float rm = red[0];
  __syncthreads();
  v0 = __expf(v0 - rm); v1 = __expf(v1 - rm);
  v2 = __expf(v2 - rm); v3 = __expf(v3 - rm);
  red[tid] = v0 + v1 + v2 + v3;
  __syncthreads();
  for (int st = 128; st > 0; st >>= 1) {
    if (tid < st) red[tid] += red[tid + st];
    __syncthreads();
  }
  float inv = 1.0f / red[0];
  p[tid] = v0 * inv; p[tid + 256] = v1 * inv;
  p[tid + 512] = v2 * inv; p[tid + 768] = v3 * inv;
}

// ----------------------------------------------------------------- P @ V
// grid (LM/16, H, B), block 128; wave w -> dv columns [w*16, w*16+16).
// V slab (32 keys x 64 dv) staged in LDS per k-step with b128 loads.
__global__ __launch_bounds__(128) void attn_pv_kernel(
    const float* __restrict__ attnW, const __bf16* __restrict__ Vp,
    __bf16* __restrict__ ctx) {
  const int b = blockIdx.z, h = blockIdx.y, q0 = blockIdx.x * 16;
  const int tid = threadIdx.x, lane = tid & 31, wave = tid >> 5;

  __shared__ __bf16 sV[32 * 64];  // [key][dv]

  const float*  Pbase = attnW + ((size_t)(b * N_HEADS + h) * LM + q0) * LP;
  const __bf16* Vblk  = Vp + (size_t)b * LP * D_MODEL + h * DK;

  v8f acc = {0.f,0.f,0.f,0.f,0.f,0.f,0.f,0.f};
  const int m   = lane & 15;
  const int kba = (lane < 16) ? 0 : 8;
  const int kbb = (lane < 16) ? 0 : 16;
  const int n   = lane & 15;
  const float* prow = Pbase + (size_t)m * LP;

  for (int ks = 0; ks < 32; ++ks) {
    int k0 = ks * 32;
    __syncthreads();
#pragma unroll
    for (int t = 0; t < 2; ++t) {  // 2 x b128 per thread = 32x64 slab
      int i = tid * 2 + t;
      int r = i >> 3;
      int c = (i & 7) * 8;
      *(uint4*)(sV + r * 64 + c) =
          *(const uint4*)(Vblk + (size_t)(k0 + r) * D_MODEL + c);
    }
    __syncthreads();

    v16bf fa, fb;
    const float2* pr = (const float2*)(prow + k0 + kba);
#pragma unroll
    for (int q = 0; q < 8; ++q) {
      float2 f2 = (q < 4) ? pr[q] : pr[q + 4];  // pairs at koff 0,2,4,6,16,18,20,22
      fa[2 * q]     = (__bf16)f2.x;
      fa[2 * q + 1] = (__bf16)f2.y;
      int key = kba + 2 * q + ((q < 4) ? 0 : 8);
      fb[2 * q]     = sV[key * 64 + wave * 16 + n];
      fb[2 * q + 1] = sV[(key + 1) * 64 + wave * 16 + n];
    }
    acc = __builtin_amdgcn_wmma_f32_16x16x32_bf16(false, fa, false, fb,
                                                  (short)0, acc, false, false);
  }
#pragma unroll
  for (int q = 0; q < 8; ++q) {
    int mr = q + ((lane < 16) ? 0 : 8);
    ctx[(size_t)(b * LM + q0 + mr) * D_MODEL + h * DK + wave * 16 + n] = (__bf16)acc[q];
  }
}

// -------------------------------------------- residual add + LayerNorm (d=768)
__global__ __launch_bounds__(256) void add_ln_kernel(
    const float* __restrict__ a, const float* __restrict__ r,
    const float* __restrict__ g, const float* __restrict__ beta,
    float* __restrict__ outF, __bf16* __restrict__ outB) {
  const int row = blockIdx.x;
  __shared__ float s[D_MODEL];
  __shared__ float red[256];
  int tid = threadIdx.x;
  float local = 0.f;
  for (int c = tid; c < D_MODEL; c += 256) {
    float v = a[(size_t)row * D_MODEL + c] + r[(size_t)row * D_MODEL + c];
    s[c] = v;
    local += v;
  }
  red[tid] = local;
  __syncthreads();
  for (int st = 128; st > 0; st >>= 1) {
    if (tid < st) red[tid] += red[tid + st];
    __syncthreads();
  }
  float mean = red[0] * (1.0f / D_MODEL);
  __syncthreads();
  local = 0.f;
  for (int c = tid; c < D_MODEL; c += 256) {
    float d0 = s[c] - mean;
    local += d0 * d0;
  }
  red[tid] = local;
  __syncthreads();
  for (int st = 128; st > 0; st >>= 1) {
    if (tid < st) red[tid] += red[tid + st];
    __syncthreads();
  }
  float inv = rsqrtf(red[0] * (1.0f / D_MODEL) + 1e-5f);
  for (int c = tid; c < D_MODEL; c += 256) {
    float y = (s[c] - mean) * inv * g[c] + beta[c];
    outF[(size_t)row * D_MODEL + c] = y;
    if (outB) outB[(size_t)row * D_MODEL + c] = (__bf16)y;
  }
}

// ---------------------------------------------------------------------------
extern "C" void kernel_launch(void* const* d_in, const int* in_sizes, int n_in,
                              void* d_out, int out_size, void* d_ws, size_t ws_size,
                              hipStream_t stream) {
  (void)in_sizes; (void)n_in; (void)out_size; (void)ws_size;
  const float* mol_h  = (const float*)d_in[0];   // (8,512,768)
  const float* prot_h = (const float*)d_in[1];   // (8,1024,768)
  const unsigned char* key_mask = (const unsigned char*)d_in[2];  // (8,1024) bool
  const float* Wq = (const float*)d_in[3];
  const float* Wk = (const float*)d_in[4];
  const float* Wv = (const float*)d_in[5];
  const float* Wo = (const float*)d_in[6];
  const float* bo = (const float*)d_in[7];
  const float* ln1_g = (const float*)d_in[8];
  const float* ln1_b = (const float*)d_in[9];
  const float* ln2_g = (const float*)d_in[10];
  const float* ln2_b = (const float*)d_in[11];
  const float* W1 = (const float*)d_in[12];
  const float* b1 = (const float*)d_in[13];
  const float* W2 = (const float*)d_in[14];
  const float* b2 = (const float*)d_in[15];

  const size_t nMol  = (size_t)BB * LM * D_MODEL;      // 3145728
  const size_t nProt = (size_t)BB * LP * D_MODEL;      // 6291456
  const size_t nW    = (size_t)D_MODEL * D_MODEL;      // 589824
  const size_t nW1   = (size_t)D_MODEL * D_FF;         // 2359296
  const size_t nFF   = (size_t)BB * LM * D_FF;         // 12582912

  float* out_x    = (float*)d_out;                     // (8,512,768)
  float* out_attn = (float*)d_out + nMol;              // (8,12,512,1024)

  // Workspace carve-up (256B aligned).
  char* base = (char*)d_ws;
  size_t off = 0;
  auto carve = [&](size_t bytes) {
    char* p = base + off;
    off += (bytes + 255) & ~(size_t)255;
    return p;
  };
  __bf16* mol_bf  = (__bf16*)carve(nMol * 2);
  __bf16* prot_bf = (__bf16*)carve(nProt * 2);
  __bf16* Wq_bf   = (__bf16*)carve(nW * 2);
  __bf16* Wk_bf   = (__bf16*)carve(nW * 2);
  __bf16* Wv_bf   = (__bf16*)carve(nW * 2);
  __bf16* Wo_bf   = (__bf16*)carve(nW * 2);
  __bf16* W1_bf   = (__bf16*)carve(nW1 * 2);
  __bf16* W2_bf   = (__bf16*)carve(nW1 * 2);
  __bf16* Qp      = (__bf16*)carve(nMol * 2);   // (B*Lm, 768) heads packed
  __bf16* Kp      = (__bf16*)carve(nProt * 2);  // (B*Lp, 768)
  __bf16* Vp      = (__bf16*)carve(nProt * 2);  // (B*Lp, 768)
  __bf16* ctx_bf  = (__bf16*)carve(nMol * 2);   // (B*Lm, 768)
  float*  attnout = (float*)carve(nMol * 4);
  float*  x_f32   = (float*)carve(nMol * 4);
  __bf16* x_bf    = (__bf16*)carve(nMol * 2);
  __bf16* ff1_bf  = (__bf16*)carve(nFF * 2);
  float*  ff2_f32 = (float*)carve(nMol * 4);

  auto cvt = [&](const float* s, __bf16* d, size_t n) {
    size_t n4 = n / 4;
    cvt_bf16_kernel<<<dim3((unsigned)((n4 + 255) / 256)), dim3(256), 0, stream>>>(
        s, d, (int)n4);
  };
  cvt(mol_h, mol_bf, nMol);
  cvt(prot_h, prot_bf, nProt);
  cvt(Wq, Wq_bf, nW);
  cvt(Wk, Wk_bf, nW);
  cvt(Wv, Wv_bf, nW);
  cvt(Wo, Wo_bf, nW);
  cvt(W1, W1_bf, nW1);
  cvt(W2, W2_bf, nW1);

  auto gemm = [&](const __bf16* A, const __bf16* B, int M, int N, int K,
                  const float* bias, int act, float* oF, __bf16* oB) {
    gemm_bf16_kernel<<<dim3(N / 64, M / 64), dim3(256), 0, stream>>>(
        A, B, M, N, K, bias, act, oF, oB);
  };

  // Projections (heads stay packed in the 768 columns).
  gemm(mol_bf,  Wq_bf, BB * LM, D_MODEL, D_MODEL, nullptr, 0, nullptr, Qp);
  gemm(prot_bf, Wk_bf, BB * LP, D_MODEL, D_MODEL, nullptr, 0, nullptr, Kp);
  gemm(prot_bf, Wv_bf, BB * LP, D_MODEL, D_MODEL, nullptr, 0, nullptr, Vp);

  // Attention: scores -> softmax (in place, in d_out) -> P@V.
  attn_scores_kernel<<<dim3(LM / 16, N_HEADS, BB), dim3(128), 0, stream>>>(
      Qp, Kp, key_mask, out_attn);
  softmax_kernel<<<dim3(BB * N_HEADS * LM), dim3(256), 0, stream>>>(out_attn);
  attn_pv_kernel<<<dim3(LM / 16, N_HEADS, BB), dim3(128), 0, stream>>>(
      out_attn, Vp, ctx_bf);

  // Output projection (+bo), LN1, FFN, LN2.
  gemm(ctx_bf, Wo_bf, BB * LM, D_MODEL, D_MODEL, bo, 0, attnout, nullptr);
  add_ln_kernel<<<dim3(BB * LM), dim3(256), 0, stream>>>(
      mol_h, attnout, ln1_g, ln1_b, x_f32, x_bf);
  gemm(x_bf, W1_bf, BB * LM, D_FF, D_MODEL, b1, 1, nullptr, ff1_bf);
  gemm(ff1_bf, W2_bf, BB * LM, D_MODEL, D_FF, b2, 0, ff2_f32, nullptr);
  add_ln_kernel<<<dim3(BB * LM), dim3(256), 0, stream>>>(
      x_f32, ff2_f32, ln2_g, ln2_b, out_x, nullptr);
}
